// MultiHeadAttention_9526237462750
// MI455X (gfx1250) — matrix operand order repaired and checked by reading, not compiled, not run on hardware
//
#include <hip/hip_runtime.h>

// ---------- types ----------
typedef __bf16 bf16;
typedef __attribute__((ext_vector_type(16))) __bf16 v16bf;
typedef __attribute__((ext_vector_type(8)))  __bf16 v8bf;
typedef __attribute__((ext_vector_type(4)))  __bf16 v4bf;
typedef __attribute__((ext_vector_type(8)))  float  v8f;
typedef __attribute__((ext_vector_type(4)))  float  v4f;

#define DEV __device__ __forceinline__

DEV bf16 f2bf(float f) {
  unsigned u = __builtin_bit_cast(unsigned, f);
  unsigned r = (u + 0x7FFFu + ((u >> 16) & 1u)) >> 16;   // round-to-nearest-even
  unsigned short s = (unsigned short)r;
  return __builtin_bit_cast(bf16, s);
}

DEV v8f zero8() {
  v8f z = {0.f, 0.f, 0.f, 0.f, 0.f, 0.f, 0.f, 0.f};
  return z;
}

DEV v16bf catbf(v8bf lo, v8bf hi) {
  v16bf r;
#pragma unroll
  for (int i = 0; i < 8; i++) { r[i] = lo[i]; r[i + 8] = hi[i]; }
  return r;
}

// ---------- CDNA5 async global->LDS copy (ASYNCcnt-tracked) ----------
// Each enabled lane copies 16B from its own global address to its own LDS
// address (per CDNA5 ISA).
DEV void async_g2l_b128(unsigned lds_addr, const void* gaddr) {
  asm volatile("global_load_async_to_lds_b128 %0, %1, off"
               :: "v"(lds_addr), "v"(gaddr)
               : "memory");
}
DEV void wait_async0() { asm volatile("s_wait_asynccnt 0" ::: "memory"); }
DEV unsigned lds_off(const void* p) { return (unsigned)(unsigned long long)p; }

// ---------- f32 -> bf16 elementwise (x, kv pre-pass) ----------
__global__ __launch_bounds__(256) void cvt_bf16(const float* __restrict__ in,
                                                bf16* __restrict__ out) {
  const size_t i = ((size_t)blockIdx.x * 256 + threadIdx.x) * 8;
  v4f a = *(const v4f*)&in[i];
  v4f b = *(const v4f*)&in[i + 4];
  v8bf o;
#pragma unroll
  for (int e = 0; e < 4; e++) { o[e] = f2bf(a[e]); o[e + 4] = f2bf(b[e]); }
  *(v8bf*)&out[i] = o;
}

// ---------- weight transpose + f32->bf16: Wt[k][n] = W[n][k] ----------
__global__ __launch_bounds__(256) void wtrans(const float* __restrict__ W,
                                              bf16* __restrict__ Wt) {
  __shared__ float tile[32][33];
  const int tx = threadIdx.x & 31, ty = threadIdx.x >> 5;      // 32 x 8
  const int nb = (blockIdx.x & 31) * 32, kb = (blockIdx.x >> 5) * 32;
#pragma unroll
  for (int j = 0; j < 4; j++)
    tile[ty + j * 8][tx] = W[(size_t)(nb + ty + j * 8) * 1024 + kb + tx];
  __syncthreads();
#pragma unroll
  for (int j = 0; j < 4; j++)
    Wt[(size_t)(kb + ty + j * 8) * 1024 + nb + tx] = f2bf(tile[tx][ty + j * 8]);
}

// ---------- GEMM: C[M=8192, N=1024] = A_bf16[M,1024] @ Wt[1024,N] ----------
// OUT_MODE: 0 = bf16 (B,H,S,64)   (Q and V)
//           1 = bf16 (B,H,64,S)   (K stored transposed for QK^T fragments)
//           2 = f32  row-major    (final output)
template <int OUT_MODE>
__global__ __launch_bounds__(256) void gemm_wmma(const bf16* __restrict__ A,
                                                 const bf16* __restrict__ Bt,
                                                 void* __restrict__ Cp) {
  constexpr int K = 1024, N = 1024;
  __shared__ bf16 As[2][128][40];   // double-buffered, padded
  __shared__ bf16 Bs[2][32][136];

  const int t = threadIdx.x;
  const int lane = t & 31, wave = t >> 5;
  const int ln = lane & 15, hf = lane >> 4;
  const int wm = wave >> 1, wn = wave & 1;               // 4 x 2 wave grid
  const int mt = blockIdx.x >> 3, nt = blockIdx.x & 7;
  const int mBase = mt * 128, nBase = nt * 128;

  auto stage = [&](int kb, int buf) {
    const int kB = kb * 32;
#pragma unroll
    for (int j = 0; j < 2; j++) {                        // A: 128 rows x 4 chunks
      int idx = t + j * 256;
      int row = idx >> 2, c = idx & 3;
      async_g2l_b128(lds_off(&As[buf][row][c * 8]),
                     A + (size_t)(mBase + row) * K + kB + c * 8);
    }
#pragma unroll
    for (int j = 0; j < 2; j++) {                        // B: 32 rows x 16 chunks
      int idx = t + j * 256;
      int row = idx >> 4, c = idx & 15;
      async_g2l_b128(lds_off(&Bs[buf][row][c * 8]),
                     Bt + (size_t)(kB + row) * N + nBase + c * 8);
    }
  };

  v8f acc[2][4];
#pragma unroll
  for (int i = 0; i < 2; i++)
#pragma unroll
    for (int j = 0; j < 4; j++) acc[i][j] = zero8();

  stage(0, 0);
  wait_async0();
  __syncthreads();

  for (int kb = 0; kb < K / 32; ++kb) {
    const int cur = kb & 1;
    if (kb + 1 < K / 32) stage(kb + 1, cur ^ 1);         // prefetch next tile

    v16bf af[2], bfv[4];
#pragma unroll
    for (int mi = 0; mi < 2; mi++) {
      const bf16* ar = &As[cur][wm * 32 + mi * 16 + ln][hf * 8];
      af[mi] = catbf(*(const v8bf*)ar, *(const v8bf*)(ar + 16));
    }
#pragma unroll
    for (int ni = 0; ni < 4; ni++) {
      for (int s = 0; s < 16; s++)
        bfv[ni][s] = Bs[cur][(s < 8 ? hf * 8 + s : 8 + hf * 8 + s)][wn * 64 + ni * 16 + ln];
    }
#pragma unroll
    for (int mi = 0; mi < 2; mi++)
#pragma unroll
      for (int ni = 0; ni < 4; ni++)
        acc[mi][ni] = __builtin_amdgcn_wmma_f32_16x16x32_bf16(
            false, af[mi], false, bfv[ni], (short)0, acc[mi][ni], false, false);

    wait_async0();
    __syncthreads();
  }

  // ---- writeback ----
#pragma unroll
  for (int mi = 0; mi < 2; mi++)
#pragma unroll
    for (int ni = 0; ni < 4; ni++)
#pragma unroll
      for (int r = 0; r < 8; r++) {
        int mg = mBase + wm * 32 + mi * 16 + r + hf * 8;
        int ng = nBase + wn * 64 + ni * 16 + ln;
        float v = acc[mi][ni][r];
        if (OUT_MODE == 0) {
          int b = mg >> 11, s = mg & 2047, h = ng >> 6, d = ng & 63;
          ((bf16*)Cp)[((((size_t)b * 16 + h) * 2048 + s) << 6) + d] = f2bf(v);
        } else if (OUT_MODE == 1) {
          int b = mg >> 11, s = mg & 2047, h = ng >> 6, d = ng & 63;
          ((bf16*)Cp)[((((size_t)b * 16 + h) * 64 + d) << 11) + s] = f2bf(v);
        } else {
          ((float*)Cp)[(size_t)mg * 1024 + ng] = v;
        }
      }
}

// ---------- causal flash attention ----------
// Q,V: (B,H,S,64) bf16 ; Kt: (B,H,64,S) bf16 ; Ctx: (B,S,1024) bf16
// Block = 8 waves x 16 q-rows = 128 q-rows; kv tiles of 32 shared via LDS.
__global__ __launch_bounds__(256) void flash_attn(const bf16* __restrict__ Q,
                                                  const bf16* __restrict__ Kt,
                                                  const bf16* __restrict__ V,
                                                  bf16* __restrict__ Ctx) {
  __shared__ bf16 Ks[2][64][40];   // K^T tile: [d][kv], padded
  __shared__ bf16 Vs[2][32][72];   // V tile:   [kv][hd], padded
  __shared__ bf16 pb[8][16][40];   // per-wave P reshape buffer

  const int t = threadIdx.x, lane = t & 31, wave = t >> 5;
  const int ln = lane & 15, hf = lane >> 4;
  const int bid = blockIdx.x;
  const int qb = bid & 15, h = (bid >> 4) & 15, b = bid >> 8;
  const int qBase = qb * 128 + wave * 16;
  const size_t bh = (size_t)b * 16 + h;

  // staging geometry (per-thread, one async b128 each for K and V)
  const int krow = t >> 2, kc = t & 3;     // 64 rows x 4 chunks
  const int vrow = t >> 3, vc = t & 7;     // 32 rows x 8 chunks
  const bf16* Kg = Kt + (bh * 64 + krow) * 2048 + kc * 8;
  const bf16* Vg = V + (bh * 2048 + vrow) * 64 + vc * 8;

  auto stage = [&](int tile, int buf) {
    const int kvB = tile * 32;
    async_g2l_b128(lds_off(&Ks[buf][krow][kc * 8]), Kg + kvB);
    async_g2l_b128(lds_off(&Vs[buf][vrow][vc * 8]), Vg + (size_t)kvB * 64);
  };

  // Q fragments for this wave's 16 rows (held for the whole kv loop)
  const bf16* qr = Q + (bh * 2048 + qBase + ln) * 64;
  v16bf qf[2];
#pragma unroll
  for (int kf = 0; kf < 2; kf++)
    qf[kf] = catbf(*(const v8bf*)(qr + kf * 32 + hf * 8),
                   *(const v8bf*)(qr + kf * 32 + 16 + hf * 8));

  v8f o[4];
#pragma unroll
  for (int i = 0; i < 4; i++) o[i] = zero8();
  float mrow[8], lrow[8];
#pragma unroll
  for (int r = 0; r < 8; r++) { mrow[r] = -__builtin_inff(); lrow[r] = 0.0f; }

  const float Cs = 0.18033688011112042f;  // (1/sqrt(64)) * log2(e)
  const int numTiles = qb * 4 + 4;        // block-uniform causal bound

  stage(0, 0);
  wait_async0();
  __syncthreads();

  for (int tt = 0; tt < numTiles; ++tt) {
    const int cur = tt & 1;
    const int kvBase = tt * 32;
    if (tt + 1 < numTiles) stage(tt + 1, cur ^ 1);       // prefetch next tile

    // scalar (wave-uniform) participation: rows of this wave touch the tile
    if (__builtin_amdgcn_readfirstlane((kvBase <= qBase + 15) ? 1 : 0)) {
      // S = Q @ K^T for a 16x32 score tile (K from LDS)
      v8f s0 = zero8(), s1 = zero8();
#pragma unroll
      for (int kf = 0; kf < 2; kf++) {
        v16bf b0, b1;
        for (int s = 0; s < 16; s++) {
          int kk = kf * 32 + (s < 8 ? hf * 8 + s : 8 + hf * 8 + s);
          b0[s] = Ks[cur][kk][ln];
          b1[s] = Ks[cur][kk][16 + ln];
        }
        s0 = __builtin_amdgcn_wmma_f32_16x16x32_bf16(false, qf[kf], false, b0,
                                                     (short)0, s0, false, false);
        s1 = __builtin_amdgcn_wmma_f32_16x16x32_bf16(false, qf[kf], false, b1,
                                                     (short)0, s1, false, false);
      }
      // causal mask on diagonal tiles
      if (kvBase + 31 > qBase) {
        const float NEG = -__builtin_inff();
#pragma unroll
        for (int r = 0; r < 8; r++) {
          int q = qBase + r + hf * 8;
          if (kvBase + ln > q)      s0[r] = NEG;
          if (kvBase + 16 + ln > q) s1[r] = NEG;
        }
      }
      // online softmax (row stats live in the 16-lane half holding that row)
#pragma unroll
      for (int r = 0; r < 8; r++) {
        float mt = fmaxf(s0[r], s1[r]);
        mt = fmaxf(mt, __shfl_xor(mt, 1, 32));
        mt = fmaxf(mt, __shfl_xor(mt, 2, 32));
        mt = fmaxf(mt, __shfl_xor(mt, 4, 32));
        mt = fmaxf(mt, __shfl_xor(mt, 8, 32));
        float mn = fmaxf(mrow[r], mt);
        float corr = exp2f((mrow[r] - mn) * Cs);
        float p0 = exp2f((s0[r] - mn) * Cs);
        float p1 = exp2f((s1[r] - mn) * Cs);
        float rs = p0 + p1;
        rs += __shfl_xor(rs, 1, 32);
        rs += __shfl_xor(rs, 2, 32);
        rs += __shfl_xor(rs, 4, 32);
        rs += __shfl_xor(rs, 8, 32);
        lrow[r] = lrow[r] * corr + rs;
        mrow[r] = mn;
        s0[r] = p0; s1[r] = p1;
#pragma unroll
        for (int ni = 0; ni < 4; ni++) o[ni][r] *= corr;
      }
      // P (C-layout) -> LDS -> A-fragment; wave-local LDS is in-order
#pragma unroll
      for (int r = 0; r < 8; r++) {
        int row = r + hf * 8;
        pb[wave][row][ln] = f2bf(s0[r]);
        pb[wave][row][16 + ln] = f2bf(s1[r]);
      }
      asm volatile("s_wait_dscnt 0" ::: "memory");
      const bf16* pr = &pb[wave][ln][hf * 8];
      v16bf pa = catbf(*(const v8bf*)pr, *(const v8bf*)(pr + 16));

      // O += P @ V (V from LDS; gathered per column into B-fragment layout)
      const bf16* vr = &Vs[cur][0][ln];
#pragma unroll
      for (int ni = 0; ni < 4; ni++) {
        v16bf bv;
        for (int s = 0; s < 16; s++)
          bv[s] = vr[(s < 8 ? hf * 8 + s : 8 + hf * 8 + s) * 72 + ni * 16];
        o[ni] = __builtin_amdgcn_wmma_f32_16x16x32_bf16(false, pa, false, bv,
                                                        (short)0, o[ni], false, false);
      }
    }

    wait_async0();        // this wave's prefetch landed in LDS
    __syncthreads();      // everyone's prefetch landed; old buffer free
  }

  // finalize: divide by row sums, store bf16 context (B,S,1024)
#pragma unroll
  for (int r = 0; r < 8; r++) {
    float inv = 1.0f / lrow[r];
#pragma unroll
    for (int ni = 0; ni < 4; ni++) o[ni][r] *= inv;
  }
#pragma unroll
  for (int ni = 0; ni < 4; ni++)
#pragma unroll
    for (int r = 0; r < 8; r++) {
      int s = qBase + r + hf * 8;
      Ctx[((size_t)b * 2048 + s) * 1024 + h * 64 + ni * 16 + ln] = f2bf(o[ni][r]);
    }
}

// ---------- host launcher ----------
extern "C" void kernel_launch(void* const* d_in, const int* in_sizes, int n_in,
                              void* d_out, int out_size, void* d_ws, size_t ws_size,
                              hipStream_t stream) {
  (void)in_sizes; (void)n_in; (void)out_size; (void)ws_size;
  const float* x  = (const float*)d_in[0];
  const float* kv = (const float*)d_in[1];
  // d_in[2] is the causal mask; causality is applied analytically in flash_attn.
  const float* Wq = (const float*)d_in[3];
  const float* Wk = (const float*)d_in[4];
  const float* Wv = (const float*)d_in[5];
  const float* Wo = (const float*)d_in[6];

  char* ws = (char*)d_ws;
  const size_t MB = 1024 * 1024;
  bf16* Wqt = (bf16*)(ws + 0 * MB);
  bf16* Wkt = (bf16*)(ws + 2 * MB);
  bf16* Wvt = (bf16*)(ws + 4 * MB);
  bf16* Wot = (bf16*)(ws + 6 * MB);
  bf16* Qb  = (bf16*)(ws + 8 * MB);   // (B,H,S,64)
  bf16* Ktb = (bf16*)(ws + 24 * MB);  // (B,H,64,S)
  bf16* Vb  = (bf16*)(ws + 40 * MB);  // (B,H,S,64)
  bf16* xb  = (bf16*)(ws + 56 * MB);  // x in bf16; region reused as Ctx below
  bf16* Ctx = (bf16*)(ws + 56 * MB);  // (B,S,1024); aliases xb (stream-ordered)
  bf16* kvb = (bf16*)(ws + 72 * MB);  // kv in bf16
  // total: 88 MB of d_ws

  // pre-passes: weights transpose+convert, activations convert
  wtrans<<<1024, 256, 0, stream>>>(Wq, Wqt);
  wtrans<<<1024, 256, 0, stream>>>(Wk, Wkt);
  wtrans<<<1024, 256, 0, stream>>>(Wv, Wvt);
  wtrans<<<1024, 256, 0, stream>>>(Wo, Wot);
  cvt_bf16<<<4096, 256, 0, stream>>>(x,  xb);
  cvt_bf16<<<4096, 256, 0, stream>>>(kv, kvb);

  // projections (WMMA, async double-buffered staging)
  gemm_wmma<0><<<512, 256, 0, stream>>>(xb,  Wqt, Qb);
  gemm_wmma<1><<<512, 256, 0, stream>>>(kvb, Wkt, Ktb);
  gemm_wmma<0><<<512, 256, 0, stream>>>(kvb, Wvt, Vb);

  // causal flash attention
  flash_attn<<<1024, 256, 0, stream>>>(Qb, Ktb, Vb, Ctx);

  // output projection -> f32
  gemm_wmma<2><<<512, 256, 0, stream>>>(Ctx, Wot, d_out);
}
